// SCN_24378234372723
// MI455X (gfx1250) — matrix operand order repaired and checked by reading, not compiled, not run on hardware
//
#include <hip/hip_runtime.h>
#include <hip/hip_bf16.h>

typedef __attribute__((ext_vector_type(16))) __bf16 v16bf;
typedef __attribute__((ext_vector_type(8)))  float  v8f;

#define B_BATCH 64
#define C_DIM   64
#define OUT_DIM 32

// -------------------------------------------------------------------------
// d[i] = rowsum(|L[i,:]|);  dinv[i] = d!=0 ? rsqrt(d) : 0
// one 256-thread block per row
// -------------------------------------------------------------------------
__global__ __launch_bounds__(256)
void scn_dinv_kernel(const float* __restrict__ L, float* __restrict__ dinv, int N)
{
    __shared__ float red[256];
    const int row = blockIdx.x;
    const float* p = L + (size_t)row * N;
    float s = 0.f;
    for (int j = threadIdx.x; j < N; j += 256) s += fabsf(p[j]);
    red[threadIdx.x] = s;
    __syncthreads();
    for (int off = 128; off > 0; off >>= 1) {
        if (threadIdx.x < off) red[threadIdx.x] += red[threadIdx.x + off];
        __syncthreads();
    }
    if (threadIdx.x == 0) {
        float d = red[0];
        dinv[row] = (d != 0.f) ? rsqrtf(d) : 0.f;
    }
}

// -------------------------------------------------------------------------
// Yt[c*N + k] = bf16( dinv[k] * sum_j x[k,j] * W[j,c] )   (column-major Y')
// W (64x64 layer slice) staged in LDS; f32 accumulate for precision.
// -------------------------------------------------------------------------
__global__ __launch_bounds__(256)
void scn_ymap_kernel(const float* __restrict__ x, const float* __restrict__ W,
                     const float* __restrict__ dinv, __bf16* __restrict__ Yt, int N)
{
    __shared__ float Ws[C_DIM * C_DIM];
    for (int i = threadIdx.x; i < C_DIM * C_DIM; i += 256) Ws[i] = W[i];
    __syncthreads();

    const int idx = blockIdx.x * 256 + threadIdx.x;
    const int k = idx >> 6;
    const int c = idx & 63;
    const float* xr = x + (size_t)k * C_DIM;
    float s = 0.f;
#pragma unroll
    for (int j = 0; j < C_DIM; ++j) s += xr[j] * Ws[j * C_DIM + c];
    Yt[(size_t)c * N + k] = (__bf16)(s * dinv[k]);
}

// -------------------------------------------------------------------------
// Big GEMM: Xout[m,c] = relu( dinv[m] * sum_k bf16(L[m,k]) * Yt[c,k] )
// 8 waves/block, wave = 16 rows x 64 cols, K-step 32, v_wmma_f32_16x16x32_bf16
// -------------------------------------------------------------------------
__global__ __launch_bounds__(256)
void scn_lgemm_kernel(const float* __restrict__ L,     // N x N row-major f32
                      const __bf16* __restrict__ Yt,   // 64 x N (col-major Y')
                      const float* __restrict__ dinv,
                      float* __restrict__ Xout,        // N x 64 row-major f32
                      int N)
{
    const int lane   = threadIdx.x & 31;
    const int wave   = threadIdx.x >> 5;          // 0..7
    const int lrow   = lane & 15;
    const int half   = lane >> 4;                 // 0 or 1
    const int rowBase = blockIdx.x * 128 + wave * 16;

    const int m = rowBase + lrow;
    const float* __restrict__ lrowp = L + (size_t)m * N;

    v8f acc[4];
#pragma unroll
    for (int t = 0; t < 4; ++t) acc[t] = (v8f){};

    for (int kb = 0; kb < N; kb += 32) {
        // ---- A fragment (16x32 bf16, contiguous k order) ----
        // lanes 0-15: K = kb+0..15 (a[0..15])
        // lanes16-31: K = kb+16..31
        const int k0 = kb + half * 16;
        v8f fa0 = *(const v8f*)(lrowp + k0);
        v8f fa1 = *(const v8f*)(lrowp + k0 + 8);
        __builtin_prefetch((const void*)(lrowp + kb + 64), 0, 0);
        v16bf a;
#pragma unroll
        for (int i = 0; i < 8; ++i) {
            a[i]     = (__bf16)fa0[i];
            a[8 + i] = (__bf16)fa1[i];
        }

        // ---- B fragments (32x16 bf16) for 4 column tiles ----
        // lane<16: VGPR r holds K{kb+2r, kb+2r+1} of column lrow (contig in Yt)
        // lane>=16: same with K offset +16
        const int kB = kb + half * 16;
#pragma unroll
        for (int t = 0; t < 4; ++t) {
            v16bf b = *(const v16bf*)(Yt + (size_t)(t * 16 + lrow) * N + kB);
            acc[t] = __builtin_amdgcn_wmma_f32_16x16x32_bf16(
                false, a, false, b, (short)0, acc[t], false, false);
        }
    }

    // ---- epilogue: row-scale by dinv, relu, store ----
    const int mo = rowBase + half * 8;            // acc[g] row = mo + g
    float* __restrict__ obase = Xout + (size_t)mo * C_DIM + lrow;
#pragma unroll
    for (int g = 0; g < 8; ++g) {
        const float dm = dinv[mo + g];
        float* orow = obase + (size_t)g * C_DIM;
#pragma unroll
        for (int t = 0; t < 4; ++t) {
            float v = acc[t][g] * dm;
            orow[t * 16] = v > 0.f ? v : 0.f;
        }
    }
}

// -------------------------------------------------------------------------
// Deterministic segment-sum: pooled[b,c] = sum_{i: bel[i]==b} x[i,c]
// thread per (b,c); bel[i] is wave-uniform (scalar load), x reads coalesced.
// -------------------------------------------------------------------------
__global__ __launch_bounds__(256)
void scn_pool_kernel(const float* __restrict__ x, const int* __restrict__ bel,
                     float* __restrict__ pooled, int N)
{
    const int idx = blockIdx.x * 256 + threadIdx.x;   // 0 .. 64*64-1
    const int b = idx >> 6;
    const int c = idx & 63;
    float s = 0.f;
    for (int i = 0; i < N; ++i) {
        if (bel[i] == b) s += x[(size_t)i * C_DIM + c];
    }
    pooled[idx] = s;
}

// -------------------------------------------------------------------------
// out[b,o] = sum_r br_r[o] + sum_r sum_c pooled_r[b,c] * Wr_r[c,o]
// -------------------------------------------------------------------------
__global__ __launch_bounds__(256)
void scn_readout_kernel(const float* __restrict__ p0, const float* __restrict__ p1,
                        const float* __restrict__ p2,
                        const float* __restrict__ Wr0, const float* __restrict__ br0,
                        const float* __restrict__ Wr1, const float* __restrict__ br1,
                        const float* __restrict__ Wr2, const float* __restrict__ br2,
                        float* __restrict__ out)
{
    const int idx = blockIdx.x * 256 + threadIdx.x;   // 0 .. 64*32-1
    const int b = idx >> 5;
    const int o = idx & 31;
    float s = br0[o] + br1[o] + br2[o];
#pragma unroll 4
    for (int c = 0; c < C_DIM; ++c) {
        const int pc = b * C_DIM + c;
        const int wc = c * OUT_DIM + o;
        s += p0[pc] * Wr0[wc] + p1[pc] * Wr1[wc] + p2[pc] * Wr2[wc];
    }
    out[idx] = s;
}

// -------------------------------------------------------------------------
extern "C" void kernel_launch(void* const* d_in, const int* in_sizes, int n_in,
                              void* d_out, int out_size, void* d_ws, size_t ws_size,
                              hipStream_t stream)
{
    // setup_inputs() order:
    // 0:x0 1:x1 2:x2 3:L0 4:L1 5:L2 6:bel0 7:bel1 8:bel2
    // 9:W0 10:W1 11:W2 12:Wr0 13:br0 14:Wr1 15:br1 16:Wr2 17:br2
    const float* xin[3] = {(const float*)d_in[0], (const float*)d_in[1], (const float*)d_in[2]};
    const float* L[3]   = {(const float*)d_in[3], (const float*)d_in[4], (const float*)d_in[5]};
    const int*   bel[3] = {(const int*)d_in[6],   (const int*)d_in[7],   (const int*)d_in[8]};
    const float* W[3]   = {(const float*)d_in[9], (const float*)d_in[10], (const float*)d_in[11]};
    const float* Wr[3]  = {(const float*)d_in[12], (const float*)d_in[14], (const float*)d_in[16]};
    const float* br[3]  = {(const float*)d_in[13], (const float*)d_in[15], (const float*)d_in[17]};

    int N[3];
    for (int r = 0; r < 3; ++r) N[r] = in_sizes[r] / C_DIM;   // 4096, 8192, 4096

    // workspace bump allocator (256B aligned)
    char* wp = (char*)d_ws;
    auto alloc = [&](size_t bytes) -> char* {
        char* ret = wp;
        wp += (bytes + 255) & ~(size_t)255;
        return ret;
    };

    float*  dinv[3];  __bf16* Yt[3];  float* xa[3];  float* xb[3];  float* pooled[3];
    for (int r = 0; r < 3; ++r) dinv[r]   = (float*)alloc((size_t)N[r] * sizeof(float));
    for (int r = 0; r < 3; ++r) Yt[r]     = (__bf16*)alloc((size_t)N[r] * C_DIM * sizeof(__bf16));
    for (int r = 0; r < 3; ++r) xa[r]     = (float*)alloc((size_t)N[r] * C_DIM * sizeof(float));
    for (int r = 0; r < 3; ++r) xb[r]     = (float*)alloc((size_t)N[r] * C_DIM * sizeof(float));
    for (int r = 0; r < 3; ++r) pooled[r] = (float*)alloc((size_t)B_BATCH * C_DIM * sizeof(float));

    // 1) normalization vectors
    for (int r = 0; r < 3; ++r)
        scn_dinv_kernel<<<N[r], 256, 0, stream>>>(L[r], dinv[r], N[r]);

    // 2) two SCN layers: Y' = dinv ∘ (x @ W);  x = relu(dinv ∘ (L @ Y'))
    const float* xcur[3] = {xin[0], xin[1], xin[2]};
    for (int layer = 0; layer < 2; ++layer) {
        float** xout = (layer == 0) ? xa : xb;
        for (int r = 0; r < 3; ++r)
            scn_ymap_kernel<<<N[r] / 4, 256, 0, stream>>>(
                xcur[r], W[r] + (size_t)layer * C_DIM * C_DIM, dinv[r], Yt[r], N[r]);
        for (int r = 0; r < 3; ++r)
            scn_lgemm_kernel<<<N[r] / 128, 256, 0, stream>>>(
                L[r], Yt[r], dinv[r], xout[r], N[r]);
        for (int r = 0; r < 3; ++r) xcur[r] = xout[r];
    }

    // 3) deterministic pooling + readout
    for (int r = 0; r < 3; ++r)
        scn_pool_kernel<<<(B_BATCH * C_DIM) / 256, 256, 0, stream>>>(
            xcur[r], bel[r], pooled[r], N[r]);

    scn_readout_kernel<<<(B_BATCH * OUT_DIM) / 256, 256, 0, stream>>>(
        pooled[0], pooled[1], pooled[2],
        Wr[0], br[0], Wr[1], br[1], Wr[2], br[2],
        (float*)d_out);

    (void)n_in; (void)out_size; (void)ws_size;
}
